// CrossViewLoss_64518998721097
// MI455X (gfx1250) — matrix operand order repaired and checked by reading, not compiled, not run on hardware
//
#include <hip/hip_runtime.h>
#include <hip/hip_bf16.h>
#include <math.h>

#define NROWS 4096
#define DDIM  256
// 1/TEMP = 2.0f

typedef _Float16 v8h  __attribute__((ext_vector_type(8)));
typedef _Float16 v16h __attribute__((ext_vector_type(16)));
typedef float    v8f  __attribute__((ext_vector_type(8)));

// ---------------------------------------------------------------------------
// Block-wide sum (blockDim.x == 256, 8 waves of 32). Returns sum to all threads.
// ---------------------------------------------------------------------------
__device__ __forceinline__ float blockReduceSum(float v) {
    __shared__ float s[8];
    const int lane = threadIdx.x & 31;
    const int wid  = threadIdx.x >> 5;
    #pragma unroll
    for (int m = 1; m < 32; m <<= 1) v += __shfl_xor(v, m, 32);
    __syncthreads();                 // protect s[] from a previous invocation
    if (lane == 0) s[wid] = v;
    __syncthreads();
    float t = s[0];
    #pragma unroll
    for (int i = 1; i < 8; i++) t += s[i];
    return t;
}

// ---------------------------------------------------------------------------
// Kernel 1: row L2-normalize emb_i/emb_j -> f16, emit pos[r] = <z_i[r], z_j[r]>,
// and zero the row/col exp-sum accumulators.
// grid = NROWS blocks, 256 threads (one thread per element of the row).
// ---------------------------------------------------------------------------
__global__ __launch_bounds__(256)
void simclr_normalize(const float* __restrict__ emb_i,
                      const float* __restrict__ emb_j,
                      _Float16* __restrict__ zi_h,
                      _Float16* __restrict__ zj_h,
                      float* __restrict__ pos,
                      float* __restrict__ rowsum,
                      float* __restrict__ colsum) {
    const int r = blockIdx.x;
    const int t = threadIdx.x;
    const int idx = r * DDIM + t;

    const float a = emb_i[idx];
    const float b = emb_j[idx];

    const float na = blockReduceSum(a * a);
    const float nb = blockReduceSum(b * b);

    const float inv_i = 1.0f / fmaxf(sqrtf(na), 1e-12f);
    const float inv_j = 1.0f / fmaxf(sqrtf(nb), 1e-12f);

    const float zi = a * inv_i;
    const float zj = b * inv_j;
    zi_h[idx] = (_Float16)zi;
    zj_h[idx] = (_Float16)zj;

    const float d = blockReduceSum(zi * zj);
    if (t == 0) {
        pos[r]    = d;
        rowsum[r] = 0.0f;
        colsum[r] = 0.0f;
    }
}

// ---------------------------------------------------------------------------
// Kernel 2: fused S = z_i @ z_j^T with exp(S/T) row/col reductions.
// Block = 256 threads (8 waves). Output tile 128x128 per block.
// Wave (waveM 0..3, waveN 0..1) owns a 32x64 region = 2x4 WMMA 16x16 tiles.
// K loop: 256 in steps of 32 using v_wmma_f32_16x16x32_f16.
// ---------------------------------------------------------------------------
__global__ __launch_bounds__(256)
void simclr_gemm_expsum(const _Float16* __restrict__ zi,
                        const _Float16* __restrict__ zj,
                        float* __restrict__ rowsum,
                        float* __restrict__ colsum) {
    const int lane  = threadIdx.x & 31;
    const int w     = threadIdx.x >> 5;
    const int waveM = w >> 1;          // 0..3
    const int waveN = w & 1;           // 0..1
    const int h     = lane >> 4;       // 0..1 : lane half
    const int l16   = lane & 15;       // 0..15

    const int rBase = blockIdx.y * 128 + waveM * 32;   // first row of wave region
    const int cBase = blockIdx.x * 128 + waveN * 64;   // first col of wave region

    // A fragment rows (M = l16 within each 16-row tile).
    const _Float16* aRow0 = zi + (size_t)(rBase + l16) * DDIM;         // M-tile 0
    const _Float16* aRow1 = aRow0 + (size_t)16 * DDIM;                 // M-tile 1

    // B fragment rows: lane l holds column n = l16; K offset h*8 (table order).
    const _Float16* bRow[4];
    #pragma unroll
    for (int j = 0; j < 4; j++)
        bRow[j] = zj + (size_t)(cBase + j * 16 + l16) * DDIM + h * 8;

    v8f acc[2][4] = {};

    #pragma unroll
    for (int k0 = 0; k0 < DDIM; k0 += 32) {
        // A fragments: two 16-byte chunks per lane, K = h*8..h*8+7 and +16.
        const _Float16* p0 = aRow0 + k0 + h * 8;
        const _Float16* p1 = aRow1 + k0 + h * 8;
        v8h lo0 = *(const v8h*)(p0);
        v8h hi0 = *(const v8h*)(p0 + 16);
        v8h lo1 = *(const v8h*)(p1);
        v8h hi1 = *(const v8h*)(p1 + 16);
        v16h a0 = __builtin_shufflevector(lo0, hi0, 0,1,2,3,4,5,6,7,8,9,10,11,12,13,14,15);
        v16h a1 = __builtin_shufflevector(lo1, hi1, 0,1,2,3,4,5,6,7,8,9,10,11,12,13,14,15);

        #pragma unroll
        for (int j = 0; j < 4; j++) {
            // B fragment: two 16-byte chunks per lane, K = h*8..h*8+7 and +16.
            v8h blo = *(const v8h*)(bRow[j] + k0);
            v8h bhi = *(const v8h*)(bRow[j] + k0 + 16);
            v16h b = __builtin_shufflevector(blo, bhi, 0,1,2,3,4,5,6,7,8,9,10,11,12,13,14,15);
            acc[0][j] = __builtin_amdgcn_wmma_f32_16x16x32_f16(
                false, a0, false, b, (short)0, acc[0][j], false, false);
            acc[1][j] = __builtin_amdgcn_wmma_f32_16x16x32_f16(
                false, a1, false, b, (short)0, acc[1][j], false, false);
        }
    }

    // ---- epilogue: e = exp(S / T) = exp(2*S), then row/col sums ----
    #pragma unroll
    for (int i = 0; i < 2; i++)
        #pragma unroll
        for (int j = 0; j < 4; j++)
            #pragma unroll
            for (int r = 0; r < 8; r++)
                acc[i][j][r] = __expf(2.0f * acc[i][j][r]);

    // Row sums: C layout -> lane group (h) shares M = r + 8*h, N = l16.
    #pragma unroll
    for (int i = 0; i < 2; i++) {
        #pragma unroll
        for (int r = 0; r < 8; r++) {
            float s = acc[i][0][r] + acc[i][1][r] + acc[i][2][r] + acc[i][3][r];
            s += __shfl_xor(s, 1, 32);
            s += __shfl_xor(s, 2, 32);
            s += __shfl_xor(s, 4, 32);
            s += __shfl_xor(s, 8, 32);
            if (l16 == 0)
                atomicAdd(&rowsum[rBase + i * 16 + h * 8 + r], s);
        }
    }

    // Column sums: fixed N = l16 within tile j; sum over all M the wave owns.
    #pragma unroll
    for (int j = 0; j < 4; j++) {
        float s = 0.0f;
        #pragma unroll
        for (int i = 0; i < 2; i++)
            #pragma unroll
            for (int r = 0; r < 8; r++)
                s += acc[i][j][r];
        s += __shfl_xor(s, 16, 32);    // combine the two lane halves (M groups)
        if (h == 0)
            atomicAdd(&colsum[cBase + j * 16 + l16], s);
    }
}

// ---------------------------------------------------------------------------
// Kernel 3: final scalar reduction.
// loss = (1/2N) * sum_r [ log(0.5*rowsum_r) + log(0.5*colsum_r) - (2/T)*pos_r ]
// ---------------------------------------------------------------------------
__global__ __launch_bounds__(256)
void simclr_finalize(const float* __restrict__ pos,
                     const float* __restrict__ rowsum,
                     const float* __restrict__ colsum,
                     float* __restrict__ out) {
    float acc = 0.0f;
    for (int i = threadIdx.x; i < NROWS; i += 256) {
        acc += logf(0.5f * rowsum[i]) + logf(0.5f * colsum[i]) - 4.0f * pos[i];
    }
    const float tot = blockReduceSum(acc);
    if (threadIdx.x == 0) out[0] = tot / (float)(2 * NROWS);
}

// ---------------------------------------------------------------------------
extern "C" void kernel_launch(void* const* d_in, const int* in_sizes, int n_in,
                              void* d_out, int out_size, void* d_ws, size_t ws_size,
                              hipStream_t stream) {
    (void)in_sizes; (void)n_in; (void)out_size; (void)ws_size;

    const float* emb_i = (const float*)d_in[0];
    const float* emb_j = (const float*)d_in[1];
    float* out = (float*)d_out;

    // Workspace carve-up (all offsets 256B-aligned):
    //   zi_h : N*D f16 = 2 MB
    //   zj_h : N*D f16 = 2 MB
    //   pos, rowsum, colsum : N f32 each = 16 KB each
    char* ws = (char*)d_ws;
    _Float16* zi_h  = (_Float16*)(ws);
    _Float16* zj_h  = (_Float16*)(ws + (size_t)NROWS * DDIM * 2);
    float*    pos    = (float*)(ws + (size_t)NROWS * DDIM * 4);
    float*    rowsum = (float*)(ws + (size_t)NROWS * DDIM * 4 + (size_t)NROWS * 4);
    float*    colsum = (float*)(ws + (size_t)NROWS * DDIM * 4 + (size_t)NROWS * 8);

    simclr_normalize<<<NROWS, 256, 0, stream>>>(emb_i, emb_j, zi_h, zj_h,
                                                pos, rowsum, colsum);

    dim3 grid(NROWS / 128, NROWS / 128);   // 32 x 32 blocks of 128x128 tiles
    simclr_gemm_expsum<<<grid, 256, 0, stream>>>(zi_h, zj_h, rowsum, colsum);

    simclr_finalize<<<1, 256, 0, stream>>>(pos, rowsum, colsum, out);
}
